// decoder_16338055594045
// MI455X (gfx1250) — matrix operand order repaired and checked by reading, not compiled, not run on hardware
//
#include <hip/hip_runtime.h>

// ---------------------------------------------------------------------------
// MI455X (gfx1250): PacConv2d -> IN+ReLU -> Conv3x3 -> IN+ReLU
// Per-tap implicit GEMMs on v_wmma_f32_16x16x32_bf16, NHWC bf16 activations,
// tap-major bf16 weights; PAC Gaussian weights applied as per-tap D-scaling.
// Weight tiles are streamed to LDS by the Tensor Data Mover (double-buffered,
// TENSORcnt-tracked) when the toolchain exposes the TDM builtin.
// ---------------------------------------------------------------------------

typedef __bf16          v16bf __attribute__((ext_vector_type(16)));
typedef float           v8f   __attribute__((ext_vector_type(8)));
typedef unsigned short  us8   __attribute__((ext_vector_type(8)));
typedef unsigned short  us16  __attribute__((ext_vector_type(16)));
typedef unsigned int    v4u   __attribute__((ext_vector_type(4)));
typedef int             v8i   __attribute__((ext_vector_type(8)));
typedef int             v4i   __attribute__((ext_vector_type(4)));

#if __has_builtin(__builtin_amdgcn_tensor_load_to_lds) && \
    __has_builtin(__builtin_amdgcn_s_wait_tensorcnt)
#define HAVE_TDM 1
#else
#define HAVE_TDM 0
#endif

static constexpr int B_  = 4;
static constexpr int C_  = 128;
static constexpr int O1  = 128;   // PAC conv out channels
static constexpr int O2  = 64;    // decode conv out channels
static constexpr int H_  = 128;
static constexpr int W_  = 128;
static constexpr int HW  = H_ * W_;      // 16384
static constexpr int K9  = C_ * 9;       // 1152
static constexpr int PIX = 64;           // pixels (M) per workgroup
static constexpr int LDB = 136;          // padded LDS row stride (ushorts)

__device__ __forceinline__ unsigned short f2bf(float f) {
  unsigned int u = __float_as_uint(f);
  u += 0x7FFFu + ((u >> 16) & 1u);       // round-to-nearest-even
  return (unsigned short)(u >> 16);
}

#if HAVE_TDM
// LDS byte offset of a __shared__ object (generic -> AS(3) -> int)
__device__ __forceinline__ unsigned lds_addr_u32(const void* p) {
  typedef __attribute__((address_space(3))) const char lds_char;
  return (unsigned)(unsigned long long)(lds_char*)p;
}

// 2D TDM load: rows x row_elems (2-byte elems) from global (row stride
// row_stride_elems) into LDS at lds_byte_addr, inserting 4 DWORDs of padding
// after every 64 DWORDs (256B row -> 272B = LDB*2 LDS stride).
__device__ __forceinline__ void tdm_load_2d(unsigned lds_byte_addr,
                                            const void* gaddr,
                                            int row_elems, int rows,
                                            int row_stride_elems) {
  unsigned long long ga = (unsigned long long)gaddr;
  v4u g0;
  g0[0] = 1u;                                        // count=1, user mode
  g0[1] = lds_byte_addr;                             // lds_addr
  g0[2] = (unsigned)ga;                              // global_addr lo
  g0[3] = (unsigned)(ga >> 32) | 0x80000000u;        // global_addr hi | type=2
  v8i g1;
  g1[0] = (1 << 16)        // data_size = 2B
        | (1 << 20)        // pad_enable
        | (5 << 22)        // pad_interval: 64 DWORDs
        | (3 << 25);       // pad_amount: 4 DWORDs
  g1[1] = (row_elems & 0xFFFF) << 16;                          // dim0 lo16
  g1[2] = ((unsigned)row_elems >> 16) | ((rows & 0xFFFF) << 16); // dim0 hi | dim1 lo
  g1[3] = ((unsigned)rows >> 16) | ((row_elems & 0xFFFF) << 16); // dim1 hi | tile_dim0
  g1[4] = rows & 0xFFFF;                                       // tile_dim1 (tile_dim2=0)
  g1[5] = row_stride_elems;                                    // dim0_stride lo32
  g1[6] = 0;
  g1[7] = 0;
  v4i z4 = (v4i){0, 0, 0, 0};
  v8i z8 = (v8i){0, 0, 0, 0, 0, 0, 0, 0};
  // 6-arg variant (clang-23 / therock headers): groups 0..3 + extra + cpol
  __builtin_amdgcn_tensor_load_to_lds(g0, g1, z4, z4, z8, 0);
}
#endif

// ---------------------------------------------------------------------------
// Weights: f32 [O][c*9+tap] -> bf16 tap-major [O][tap*128+c]
// ---------------------------------------------------------------------------
__global__ void cvt_w_tapmajor_kernel(const float* __restrict__ in,
                                      unsigned short* __restrict__ out, int total) {
  int i = blockIdx.x * 256 + threadIdx.x;
  if (i >= total) return;
  int n = i / K9, k = i - n * K9;
  int tap = k >> 7, c = k & 127;
  out[i] = f2bf(in[n * K9 + c * 9 + tap]);
}

// ---------------------------------------------------------------------------
// NCHW f32 -> NHWC bf16, optionally fused InstanceNorm + ReLU.
// ---------------------------------------------------------------------------
template <bool NORM>
__global__ __launch_bounds__(256)
void nhwc_bf16_kernel(const float* __restrict__ src,
                      const float* __restrict__ st,
                      unsigned short* __restrict__ dst) {
  __shared__ float smu[C_], srs[C_];
  int p = blockIdx.x * 256 + threadIdx.x;   // pixel id over B*HW
  int b = p >> 14, hw = p & 16383;
  if constexpr (NORM) {
    for (int c = threadIdx.x; c < C_; c += 256) {
      smu[c] = st[2 * (b * C_ + c)];
      srs[c] = st[2 * (b * C_ + c) + 1];
    }
    __syncthreads();
  }
  const float* sp = src + ((size_t)b << 21) + hw;  // channel stride = HW
  unsigned short* dp = dst + ((size_t)p << 7);
  for (int c0 = 0; c0 < C_; c0 += 8) {
    us8 v;
#pragma unroll
    for (int j = 0; j < 8; ++j) {
      float f = sp[(size_t)(c0 + j) << 14];
      if constexpr (NORM) f = fmaxf((f - smu[c0 + j]) * srs[c0 + j], 0.f);
      v[j] = f2bf(f);
    }
    *(us8*)(dp + c0) = v;
  }
}

// ---------------------------------------------------------------------------
// Pixel-adaptive Gaussian kernel: kern[b][tap][h][w] = exp(-0.5*sum_c diff^2)
// ---------------------------------------------------------------------------
__global__ __launch_bounds__(256)
void pac_kern_kernel(const float* __restrict__ guide, float* __restrict__ kern) {
  int p = blockIdx.x * 256 + threadIdx.x;
  if (p >= B_ * HW) return;
  int b = p >> 14, hw = p & 16383;
  int h = hw >> 7, w = hw & 127;
  float acc[9];
#pragma unroll
  for (int t = 0; t < 9; ++t) acc[t] = 0.f;
  const float* gb = guide + ((size_t)b << 21);
  for (int c = 0; c < C_; ++c) {
    const float* gc = gb + (c << 14);
    float ctr = gc[hw];
#pragma unroll
    for (int t = 0; t < 9; ++t) {
      int y = h + t / 3 - 1, x = w + t % 3 - 1;
      float g = ((unsigned)y < (unsigned)H_ && (unsigned)x < (unsigned)W_)
                    ? gc[(y << 7) + x] : 0.f;
      float d = g - ctr;
      acc[t] = fmaf(d, d, acc[t]);
    }
  }
#pragma unroll
  for (int t = 0; t < 9; ++t)
    kern[(b * 9 + t) * HW + hw] = __expf(-0.5f * acc[t]);
}

// ---------------------------------------------------------------------------
// Per-tap implicit GEMM conv, WMMA bf16, f32 accumulate.
//   aT  : NHWC bf16 activations [b][y][x][c]
//   Wt  : bf16 weights [O][tap*128 + c]
//   PAC : per-tap D-scale by kern[b][tap][pixel]
// Workgroup = 8 waves = 64 pixels x O outputs. Per tap: A(64x128) staged by
// vector copies (handles x-border zeros); B(Ox128) staged by TDM into a
// double buffer (prefetch tap+1 while computing tap), else vector copies.
// ---------------------------------------------------------------------------
template <int O, bool PAC>
__global__ __launch_bounds__(256)
void gemm_wmma_kernel(const unsigned short* __restrict__ aT,
                      const float* __restrict__ kern,
                      const unsigned short* __restrict__ Wt,
                      const float* __restrict__ bias,
                      float* __restrict__ out) {
  constexpr int NT   = O / 16;         // 8 (PAC) or 4 (decode)
  constexpr int TPW  = (4 * NT) / 8;   // tiles per wave: 4 or 2 (all same mt)
  constexpr int KSZ  = PAC ? PIX * 12 : 1;
  constexpr int BUFS = HAVE_TDM ? 2 : 1;

  __shared__ __align__(16) unsigned short Ald[PIX * LDB];
  __shared__ __align__(16) unsigned short Bld[BUFS][O * LDB];
  __shared__ float Kld[KSZ];

  const int tid  = threadIdx.x;
  const int lane = tid & 31;
  const int wv   = tid >> 5;
  const int pix0 = blockIdx.x * PIX;
  const int b    = pix0 >> 14;         // uniform per block
  const int hw0  = pix0 & 16383;
  const int h    = hw0 >> 7;           // 64-pixel block lies in one image row
  const int w0   = hw0 & 127;

  if constexpr (PAC) {                 // kern tile: 64 pixels x 9 taps
    for (int e = tid; e < PIX * 9; e += 256) {
      int m = e / 9, tap = e - m * 9;
      Kld[m * 12 + tap] = kern[(b * 9 + tap) * HW + hw0 + m];
    }
  }

  v8f acc[TPW], tacc[TPW];
#pragma unroll
  for (int t = 0; t < TPW; ++t) acc[t] = (v8f){0, 0, 0, 0, 0, 0, 0, 0};

  const int half = (lane >> 4) * 8;    // A fragment K sub-offset
  const int koff = (lane >> 4) * 16;   // unused; B fragment also uses half
  const int row  = lane & 15;
  const int mt   = (wv * TPW) / NT;    // single M-tile per wave
  const int sq   = tid & 15;           // A staging column group (8 channels)

  // valid taps form a contiguous range (y bounds only trim front/back rows)
  const int tlo = (h == 0) ? 3 : 0;
  const int thi = (h == H_ - 1) ? 6 : 9;

#if HAVE_TDM
  unsigned bldaddr[2] = {lds_addr_u32(&Bld[0][0]), lds_addr_u32(&Bld[1][0])};
  if (wv == 0)                         // kick off first weight tile
    tdm_load_2d(bldaddr[tlo & 1], Wt + tlo * 128, 128, O, K9);
#endif

  for (int tap = tlo; tap < thi; ++tap) {
    const int dx = tap % 3 - 1;
    const int y  = h + tap / 3 - 1;

    __syncthreads();                   // previous tap's LDS reads done
#if HAVE_TDM
    if (wv == 0 && tap + 1 < thi)      // prefetch next tap's weights
      tdm_load_2d(bldaddr[(tap + 1) & 1], Wt + (tap + 1) * 128, 128, O, K9);
#endif
    // ---- stage A: 64 rows x 128 ch bf16 (pure 16B copies) ----
    {
      const unsigned short* base =
          aT + ((((b << 14) + (y << 7)) << 7)) + sq * 8;
#pragma unroll
      for (int it = 0; it < 4; ++it) {
        int m = (tid >> 4) + 16 * it;
        int x = w0 + m + dx;
        us8 v = (us8){0, 0, 0, 0, 0, 0, 0, 0};
        if ((unsigned)x < (unsigned)W_) v = *(const us8*)(base + (x << 7));
        *(us8*)&Ald[m * LDB + sq * 8] = v;
      }
    }
#if HAVE_TDM
    if (wv == 0) {                     // current tile landed; next may fly
      if (tap + 1 < thi) __builtin_amdgcn_s_wait_tensorcnt(1);
      else               __builtin_amdgcn_s_wait_tensorcnt(0);
    }
#else
    {
      const unsigned short* wb = Wt + tap * 128;
      for (int vv = tid; vv < O * 16; vv += 256) {
        int n = vv >> 4, q = vv & 15;
        *(us8*)&Bld[0][n * LDB + q * 8] = *(const us8*)(wb + n * K9 + q * 8);
      }
    }
    if (tap + 1 < thi)
      __builtin_prefetch(Wt + (tap + 1) * 128 + (wv * (O / 8)) * K9, 0, 1);
#endif
    __syncthreads();

    const unsigned short* bb = Bld[tap & (BUFS - 1)];
    v8f* pa = PAC ? tacc : acc;
    if constexpr (PAC) {
#pragma unroll
      for (int t = 0; t < TPW; ++t) tacc[t] = (v8f){0, 0, 0, 0, 0, 0, 0, 0};
    }
    // ---- 4 K-chunks of 32, barrier-free ----
#pragma unroll
    for (int cc = 0; cc < 4; ++cc) {
      union { us8 p[2]; us16 v; } au;
      au.p[0] = *(const us8*)&Ald[(mt * 16 + row) * LDB + cc * 32 + half];
      au.p[1] = *(const us8*)&Ald[(mt * 16 + row) * LDB + cc * 32 + 16 + half];
      v16bf a = __builtin_bit_cast(v16bf, au.v);
#pragma unroll
      for (int t = 0; t < TPW; ++t) {
        int nt = (wv * TPW + t) % NT;
        union { us8 p[2]; us16 v; } bu;
        bu.p[0] = *(const us8*)&bb[(nt * 16 + row) * LDB + cc * 32 + half];
        bu.p[1] = *(const us8*)&bb[(nt * 16 + row) * LDB + cc * 32 + 16 + half];
        v16bf bm = __builtin_bit_cast(v16bf, bu.v);
        pa[t] = __builtin_amdgcn_wmma_f32_16x16x32_bf16(
                    false, a, false, bm, (short)0, pa[t], false, false);
      }
    }
    // ---- PAC: scale this tap's partial by kern row values ----
    if constexpr (PAC) {
      float s[8];
#pragma unroll
      for (int r = 0; r < 8; ++r)
        s[r] = Kld[(mt * 16 + r + ((lane >> 4) << 3)) * 12 + tap];
#pragma unroll
      for (int t = 0; t < TPW; ++t)
#pragma unroll
        for (int r = 0; r < 8; ++r)
          acc[t][r] = fmaf(tacc[t][r], s[r], acc[t][r]);
    }
  }

  // ---- store D (+bias) to NCHW f32; col = lane&15, row = r + 8*(lane>>4) ----
#pragma unroll
  for (int t = 0; t < TPW; ++t) {
    int nt = (wv * TPW + t) % NT;
    int n  = nt * 16 + (lane & 15);
    float bv = bias[n];
#pragma unroll
    for (int r = 0; r < 8; ++r) {
      int m = mt * 16 + r + ((lane >> 4) << 3);
      out[((b * O + n) << 14) + hw0 + m] = acc[t][r] + bv;
    }
  }
}

// ---------------------------------------------------------------------------
// InstanceNorm stats: one block per (b,channel); biased variance.
// ---------------------------------------------------------------------------
__global__ __launch_bounds__(256)
void instnorm_stats_kernel(const float* __restrict__ x, float* __restrict__ st) {
  __shared__ float ss[256], sq[256];
  const float* p = x + ((size_t)blockIdx.x << 14);
  float s = 0.f, q = 0.f;
  for (int j = threadIdx.x; j < HW; j += 256) {
    float v = p[j];
    s += v;
    q = fmaf(v, v, q);
  }
  ss[threadIdx.x] = s;
  sq[threadIdx.x] = q;
  __syncthreads();
  for (int stn = 128; stn > 0; stn >>= 1) {
    if ((int)threadIdx.x < stn) {
      ss[threadIdx.x] += ss[threadIdx.x + stn];
      sq[threadIdx.x] += sq[threadIdx.x + stn];
    }
    __syncthreads();
  }
  if (threadIdx.x == 0) {
    float mu  = ss[0] * (1.f / HW);
    float var = sq[0] * (1.f / HW) - mu * mu;
    st[2 * blockIdx.x]     = mu;
    st[2 * blockIdx.x + 1] = rsqrtf(var + 1e-5f);
  }
}

// normalize + ReLU -> f32 (final output, NCHW)
__global__ __launch_bounds__(256)
void norm_relu_f32_kernel(const float* __restrict__ x,
                          const float* __restrict__ st,
                          float* __restrict__ out, int total) {
  int i = (blockIdx.x * 256 + threadIdx.x) * 4;
  if (i >= total) return;
  int ch = i >> 14;
  float mu = st[2 * ch], rs = st[2 * ch + 1];
  float4 v = *(const float4*)(x + i);
  float4 o;
  o.x = fmaxf((v.x - mu) * rs, 0.f);
  o.y = fmaxf((v.y - mu) * rs, 0.f);
  o.z = fmaxf((v.z - mu) * rs, 0.f);
  o.w = fmaxf((v.w - mu) * rs, 0.f);
  *(float4*)(out + i) = o;
}

// ---------------------------------------------------------------------------
extern "C" void kernel_launch(void* const* d_in, const int* in_sizes, int n_in,
                              void* d_out, int out_size, void* d_ws, size_t ws_size,
                              hipStream_t stream) {
  const float* feat   = (const float*)d_in[0];
  const float* guide  = (const float*)d_in[1];
  const float* pac_w  = (const float*)d_in[2];
  const float* pac_b  = (const float*)d_in[3];
  const float* conv_w = (const float*)d_in[4];
  const float* conv_b = (const float*)d_in[5];
  float* out = (float*)d_out;

  char* ws = (char*)d_ws;
  size_t off = 0;
  float* kern = (float*)(ws + off);                    off += (size_t)B_ * 9 * HW * 4;
  unsigned short* featT = (unsigned short*)(ws + off); off += (size_t)B_ * HW * C_ * 2;
  unsigned short* wpac  = (unsigned short*)(ws + off); off += (size_t)O1 * K9 * 2;
  unsigned short* wdec  = (unsigned short*)(ws + off); off += (size_t)O2 * K9 * 2;
  float* pac  = (float*)(ws + off);                    off += (size_t)B_ * O1 * HW * 4;
  unsigned short* xbf = (unsigned short*)(ws + off);   off += (size_t)B_ * C_ * HW * 2;
  float* ybuf = (float*)(ws + off);                    off += (size_t)B_ * O2 * HW * 4;
  float* st1  = (float*)(ws + off);                    off += (size_t)B_ * O1 * 2 * 4;
  float* st2  = (float*)(ws + off);                    off += (size_t)B_ * O2 * 2 * 4;

  // 1) weights -> bf16, tap-major
  cvt_w_tapmajor_kernel<<<(O1 * K9) / 256, 256, 0, stream>>>(pac_w, wpac, O1 * K9);
  cvt_w_tapmajor_kernel<<<(O2 * K9) / 256, 256, 0, stream>>>(conv_w, wdec, O2 * K9);

  // 2) feat -> NHWC bf16 ; Gaussian kernel weights from guide
  nhwc_bf16_kernel<false><<<(B_ * HW) / 256, 256, 0, stream>>>(feat, nullptr, featT);
  pac_kern_kernel<<<(B_ * HW) / 256, 256, 0, stream>>>(guide, kern);

  // 3) PAC conv: 9 per-tap GEMMs (65536 x 128 -> 128) + kern scale + bias
  gemm_wmma_kernel<O1, true><<<(B_ * HW) / PIX, 256, 0, stream>>>(
      featT, kern, wpac, pac_b, pac);

  // 4) InstanceNorm + ReLU -> NHWC bf16
  instnorm_stats_kernel<<<B_ * O1, 256, 0, stream>>>(pac, st1);
  nhwc_bf16_kernel<true><<<(B_ * HW) / 256, 256, 0, stream>>>(pac, st1, xbf);

  // 5) decode conv: 9 per-tap GEMMs (65536 x 128 -> 64) + bias
  gemm_wmma_kernel<O2, false><<<(B_ * HW) / PIX, 256, 0, stream>>>(
      xbf, nullptr, wdec, conv_b, ybuf);

  // 6) InstanceNorm + ReLU -> f32 output
  instnorm_stats_kernel<<<B_ * O2, 256, 0, stream>>>(ybuf, st2);
  norm_relu_f32_kernel<<<(B_ * O2 * HW) / 1024, 256, 0, stream>>>(
      ybuf, st2, out, B_ * O2 * HW);
}
